// MultiHeadSelfAttention_37787122270879
// MI455X (gfx1250) — matrix operand order repaired and checked by reading, not compiled, not run on hardware
//
#include <hip/hip_runtime.h>
#include <hip/hip_bf16.h>

typedef unsigned short ushort_t;
typedef __attribute__((ext_vector_type(16))) __bf16 v16bf;
typedef __attribute__((ext_vector_type(8)))  float  v8f;

#define Bb  4
#define Ss  2048
#define Dd  1024
#define Hh  16
#define DKk 64

// ---------- helpers ----------

__device__ __forceinline__ ushort_t f2bf(float x) {
  union { float f; unsigned u; } v; v.f = x;
  unsigned r = v.u + 0x7FFFu + ((v.u >> 16) & 1u);   // round to nearest even
  return (ushort_t)(r >> 16);
}

// build a v16bf A/B fragment from two 16-byte chunks (8 bf16 each)
__device__ __forceinline__ v16bf ld2x16(const ushort_t* p0, const ushort_t* p1) {
  union { uint4 u[2]; v16bf v; } t;
  t.u[0] = *reinterpret_cast<const uint4*>(p0);
  t.u[1] = *reinterpret_cast<const uint4*>(p1);
  return t.v;
}

__device__ __forceinline__ v8f wmma_bf16(v16bf a, v16bf b, v8f c) {
  return __builtin_amdgcn_wmma_f32_16x16x32_bf16(
      /*neg_a=*/false, a, /*neg_b=*/false, b,
      /*c_mod=*/(short)0, c, /*reuse_a=*/false, /*reuse_b=*/false);
}

// async copy 16 bytes: global -> LDS (tracked by ASYNCcnt)
__device__ __forceinline__ void async_g2l_b128(unsigned lds_off, const void* gptr) {
  asm volatile("global_load_async_to_lds_b128 %0, %1, off"
               :: "v"(lds_off), "v"((unsigned long long)gptr) : "memory");
}

// reductions within a 16-lane half (lanes 0-15 / 16-31 stay separate)
__device__ __forceinline__ float hmax16(float v) {
#pragma unroll
  for (int m = 1; m <= 8; m <<= 1) v = fmaxf(v, __shfl_xor(v, m, 32));
  return v;
}
__device__ __forceinline__ float hadd16(float v) {
#pragma unroll
  for (int m = 1; m <= 8; m <<= 1) v += __shfl_xor(v, m, 32);
  return v;
}

// ---------- packing kernels ----------

__global__ __launch_bounds__(256)
void pack_bf16_kernel(const float* __restrict__ in, ushort_t* __restrict__ out, int n) {
  int i = blockIdx.x * blockDim.x + threadIdx.x;
  if (i < n) out[i] = f2bf(in[i]);
}

// WT[k*Dd + n] = W[n*Dd + k]   (so y = x @ W^T becomes A[M,K] * WT[K,N])
__global__ __launch_bounds__(256)
void transpose_pack_kernel(const float* __restrict__ W, ushort_t* __restrict__ WT) {
  int i = blockIdx.x * blockDim.x + threadIdx.x;     // i = k*Dd + n
  int k = i >> 10;
  int n = i & 1023;
  WT[i] = f2bf(W[n * Dd + k]);
}

// ---------- bf16 WMMA GEMM:  C[M,N] = A[M,K] * Bt[K,N]  (fp32 out) ----------
// block = 128 threads = 4 waves; wave tile 32(M) x 64(N); block tile 128 x 64
// B tile (32x64 bf16, shared by all 4 waves) staged in LDS via async copy,
// double buffered; A fragments loaded directly (no cross-wave reuse).

__global__ __launch_bounds__(128)
void gemm_bf16_kernel(const ushort_t* __restrict__ A, const ushort_t* __restrict__ Bt,
                      float* __restrict__ C, int M, int N, int K) {
  __shared__ ushort_t sB[2][32][64];   // 2 x 4KB double-buffered B tile

  const int w    = threadIdx.x >> 5;
  const int lane = threadIdx.x & 31;
  const int l    = lane & 15;
  const int hi   = lane >> 4;
  const int tid  = threadIdx.x;

  const int mb = blockIdx.y * 128 + w * 32;
  const int nb = blockIdx.x * 64;

  v8f acc[2][4];
#pragma unroll
  for (int m = 0; m < 2; ++m)
#pragma unroll
    for (int n = 0; n < 4; ++n) acc[m][n] = (v8f){0.f,0.f,0.f,0.f,0.f,0.f,0.f,0.f};

  const int KT = K >> 5;               // number of 32-wide k tiles

  // issue async B-tile t into buffer b: 256 chunks of 16B, 2 per thread
  auto issueB = [&](int t, int b) {
#pragma unroll
    for (int j = 0; j < 2; ++j) {
      const int c    = tid + j * 128;
      const int row  = c >> 3;         // 0..31  (k within tile)
      const int colc = c & 7;          // 0..7   (8-ushort chunk)
      const ushort_t* g = Bt + (size_t)(t * 32 + row) * N + nb + colc * 8;
      const unsigned lds = (unsigned)(uintptr_t)&sB[b][row][colc * 8];
      async_g2l_b128(lds, g);
    }
  };

  issueB(0, 0);                        // prologue

  for (int t = 0; t < KT; ++t) {
    const int cur = t & 1;
    const bool have_next = (t + 1 < KT);
    if (have_next) issueB(t + 1, cur ^ 1);

    // async loads retire in order: waiting for <=2 outstanding means the
    // older pair (tile t) has landed; <=0 on the last iteration.
    if (have_next) asm volatile("s_wait_asynccnt 0x2" ::: "memory");
    else           asm volatile("s_wait_asynccnt 0x0" ::: "memory");
    __syncthreads();                   // all waves' tile-t chunks visible

    // A fragments (direct global, per-wave rows)
    v16bf af[2];
#pragma unroll
    for (int m = 0; m < 2; ++m) {
      const ushort_t* ar = A + (size_t)(mb + m * 16 + l) * K + t * 32;
      af[m] = ld2x16(ar + hi * 8, ar + 16 + hi * 8);
    }
    // B fragments from LDS (lane = n column, elements = k in table order)
    v16bf bfr[4];
#pragma unroll
    for (int n = 0; n < 4; ++n)
#pragma unroll
      for (int s = 0; s < 16; ++s)
        bfr[n][s] = __builtin_bit_cast(__bf16, sB[cur][(s & 7) + hi * 8 + (s >> 3) * 16][n * 16 + l]);

#pragma unroll
    for (int m = 0; m < 2; ++m)
#pragma unroll
      for (int n = 0; n < 4; ++n) acc[m][n] = wmma_bf16(af[m], bfr[n], acc[m][n]);

    __syncthreads();                   // reads done before buffer is rewritten
  }

#pragma unroll
  for (int m = 0; m < 2; ++m)
#pragma unroll
    for (int n = 0; n < 4; ++n)
#pragma unroll
      for (int r = 0; r < 8; ++r) {
        int row = mb + m * 16 + r + 8 * hi;
        int col = nb + n * 16 + l;
        C[(size_t)row * N + col] = acc[m][n][r];
      }
}

// ---------- RoPE + repack ----------
// Qf/Kf/Vf: fp32 [B*S, D] rows (b*S+s), cols h*64+d
// Qb: bf16 [b,h,s,dk] (pre-scaled by 1/8);  Ktb: bf16 [b,h,dk,s];  Vb: bf16 [b,h,s,dk]

__global__ __launch_bounds__(64)
void rope_pack_kernel(const float* __restrict__ Qf, const float* __restrict__ Kf,
                      const float* __restrict__ Vf,
                      ushort_t* __restrict__ Qb, ushort_t* __restrict__ Ktb,
                      ushort_t* __restrict__ Vb) {
  const int s = blockIdx.x, h = blockIdx.y, b = blockIdx.z;
  const int t = threadIdx.x;
  const size_t rowf = ((size_t)b * Ss + s) * Dd + h * DKk;
  const size_t bh   = (size_t)b * Hh + h;

  // V copy (all 64 lanes)
  Vb[(bh * Ss + s) * DKk + t] = f2bf(Vf[rowf + t]);

  if (t < 32) {
    const int d0 = 2 * t, d1 = 2 * t + 1;
    const float inv_freq = exp2f(-((float)d0 / (float)DKk) * 13.287712379549449f); // log2(10000)
    const float ang = (float)s * inv_freq;
    float sn, cs;
    __sincosf(ang, &sn, &cs);

    float q0v = Qf[rowf + d0], q1v = Qf[rowf + d1];
    float k0v = Kf[rowf + d0], k1v = Kf[rowf + d1];

    float qr0 = (q0v * cs - q1v * sn) * 0.125f;   // fold 1/sqrt(64)
    float qr1 = (q0v * sn + q1v * cs) * 0.125f;
    float kr0 = k0v * cs - k1v * sn;
    float kr1 = k0v * sn + k1v * cs;

    Qb[(bh * Ss + s) * DKk + d0] = f2bf(qr0);
    Qb[(bh * Ss + s) * DKk + d1] = f2bf(qr1);
    Ktb[(bh * DKk + d0) * Ss + s] = f2bf(kr0);
    Ktb[(bh * DKk + d1) * Ss + s] = f2bf(kr1);
  }
}

// ---------- flash attention (causal) ----------
// grid (S/64, H, B); 128 threads = 4 waves; wave w handles q rows [q0, q0+16)
// Of: bf16 [b, s, h*64+d]  (head-concat layout for final projection)

__global__ __launch_bounds__(128)
void attn_kernel(const ushort_t* __restrict__ Q, const ushort_t* __restrict__ Kt,
                 const ushort_t* __restrict__ V, ushort_t* __restrict__ O) {
  __shared__ ushort_t Pld[4][16][40];   // per-wave 16x32 P tile (+pad)

  const int w    = threadIdx.x >> 5;
  const int lane = threadIdx.x & 31;
  const int l    = lane & 15;
  const int hi   = lane >> 4;
  const int h = blockIdx.y, b = blockIdx.z;
  const int q0 = blockIdx.x * 64 + w * 16;

  const size_t bh = (size_t)b * Hh + h;
  const ushort_t* Qh  = Q  + bh * Ss * DKk;
  const ushort_t* Kth = Kt + bh * DKk * Ss;
  const ushort_t* Vh  = V  + bh * Ss * DKk;

  // Q A-fragments: a0 covers d 0..31, a1 covers d 32..63
  const ushort_t* qrow = Qh + (size_t)(q0 + l) * DKk;
  const v16bf a0 = ld2x16(qrow + hi * 8,      qrow + 16 + hi * 8);
  const v16bf a1 = ld2x16(qrow + 32 + hi * 8, qrow + 48 + hi * 8);

  const v8f vz = (v8f){0.f,0.f,0.f,0.f,0.f,0.f,0.f,0.f};
  v8f o0 = vz, o1 = vz, o2 = vz, o3 = vz;
  float mx[8], sm[8];
#pragma unroll
  for (int r = 0; r < 8; ++r) { mx[r] = -__builtin_inff(); sm[r] = 0.f; }

  const int ntiles = (q0 + 16 + 31) >> 5;
  for (int tIt = 0; tIt < ntiles; ++tIt) {
    const int c = tIt * 32;

    // K^T B-fragments (lane = key column, elements = d in table order)
    v16bf kb[4];
#pragma unroll
    for (int j = 0; j < 4; ++j)
#pragma unroll
      for (int s = 0; s < 16; ++s)
        kb[j][s] = __builtin_bit_cast(__bf16, Kth[(size_t)((j & 1) * 32 + (s & 7) + hi * 8 + (s >> 3) * 16) * Ss + c + (j >> 1) * 16 + l]);
    v8f s0 = wmma_bf16(a0, kb[0], vz);
    s0     = wmma_bf16(a1, kb[1], s0);
    v8f s1 = wmma_bf16(a0, kb[2], vz);
    s1     = wmma_bf16(a1, kb[3], s1);

    if (c + 31 > q0) {            // diagonal tile: causal mask
#pragma unroll
      for (int r = 0; r < 8; ++r) {
        const int row = q0 + r + 8 * hi;
        if (c + l > row)      s0[r] = -__builtin_inff();
        if (c + 16 + l > row) s1[r] = -__builtin_inff();
      }
    }

    // online softmax (per row r within this half-wave)
#pragma unroll
    for (int r = 0; r < 8; ++r) {
      const float cand = hmax16(fmaxf(s0[r], s1[r]));
      const float mnew = fmaxf(mx[r], cand);
      const float alpha = __expf(mx[r] - mnew);
      const float p0 = __expf(s0[r] - mnew);
      const float p1 = __expf(s1[r] - mnew);
      sm[r] = sm[r] * alpha + hadd16(p0 + p1);
      mx[r] = mnew;
      o0[r] *= alpha; o1[r] *= alpha; o2[r] *= alpha; o3[r] *= alpha;
      const int rr = r + 8 * hi;
      Pld[w][rr][l]      = f2bf(p0);
      Pld[w][rr][16 + l] = f2bf(p1);
    }

    // per-wave LDS fence: DS ops of one wave stay in order; wait writes done
    asm volatile("s_wait_dscnt 0x0" ::: "memory");

    // reload P in A-fragment layout (transpose via LDS)
    const v16bf ap = ld2x16(&Pld[w][l][hi * 8], &Pld[w][l][16 + hi * 8]);

    // V B-fragments (lane = d column, elements = key in table order)
    v16bf vb[4];
#pragma unroll
    for (int j = 0; j < 4; ++j)
#pragma unroll
      for (int s = 0; s < 16; ++s)
        vb[j][s] = __builtin_bit_cast(__bf16, Vh[(size_t)(c + (s & 7) + hi * 8 + (s >> 3) * 16) * DKk + j * 16 + l]);
    o0 = wmma_bf16(ap, vb[0], o0);
    o1 = wmma_bf16(ap, vb[1], o1);
    o2 = wmma_bf16(ap, vb[2], o2);
    o3 = wmma_bf16(ap, vb[3], o3);
  }

#pragma unroll
  for (int r = 0; r < 8; ++r) {
    const float inv = 1.0f / sm[r];
    const int qrow_g = q0 + r + 8 * hi;
    ushort_t* op = O + ((size_t)b * Ss + qrow_g) * Dd + h * DKk;
    op[l]      = f2bf(o0[r] * inv);
    op[16 + l] = f2bf(o1[r] * inv);
    op[32 + l] = f2bf(o2[r] * inv);
    op[48 + l] = f2bf(o3[r] * inv);
  }
}

// ---------- host launch ----------

extern "C" void kernel_launch(void* const* d_in, const int* in_sizes, int n_in,
                              void* d_out, int out_size, void* d_ws, size_t ws_size,
                              hipStream_t stream) {
  const float* x  = (const float*)d_in[0];
  const float* Wq = (const float*)d_in[1];
  const float* Wk = (const float*)d_in[2];
  const float* Wv = (const float*)d_in[3];
  const float* Wo = (const float*)d_in[4];

  const size_t MB = 1024ull * 1024ull;
  char* ws = (char*)d_ws;
  ushort_t* Xb  = (ushort_t*)(ws);                 // 16 MB
  ushort_t* WqT = (ushort_t*)(ws + 16 * MB);       //  2 MB each
  ushort_t* WkT = (ushort_t*)(ws + 18 * MB);
  ushort_t* WvT = (ushort_t*)(ws + 20 * MB);
  ushort_t* WoT = (ushort_t*)(ws + 22 * MB);
  float*    Qf  = (float*)(ws + 24 * MB);          // 32 MB each
  float*    Kf  = (float*)(ws + 56 * MB);
  float*    Vf  = (float*)(ws + 88 * MB);
  ushort_t* Qb  = (ushort_t*)(ws + 120 * MB);      // 16 MB each
  ushort_t* Ktb = (ushort_t*)(ws + 136 * MB);
  ushort_t* Vb  = (ushort_t*)(ws + 152 * MB);
  ushort_t* Of  = (ushort_t*)(ws + 168 * MB);      // 16 MB

  const int M = Bb * Ss;          // 8192
  const int nElem = M * Dd;       // 8,388,608
  const int wElem = Dd * Dd;      // 1,048,576

  pack_bf16_kernel<<<(nElem + 255) / 256, 256, 0, stream>>>(x, Xb, nElem);
  transpose_pack_kernel<<<wElem / 256, 256, 0, stream>>>(Wq, WqT);
  transpose_pack_kernel<<<wElem / 256, 256, 0, stream>>>(Wk, WkT);
  transpose_pack_kernel<<<wElem / 256, 256, 0, stream>>>(Wv, WvT);
  transpose_pack_kernel<<<wElem / 256, 256, 0, stream>>>(Wo, WoT);

  dim3 ggrid(Dd / 64, M / 128);   // (16, 64)
  gemm_bf16_kernel<<<ggrid, 128, 0, stream>>>(Xb, WqT, Qf, M, Dd, Dd);
  gemm_bf16_kernel<<<ggrid, 128, 0, stream>>>(Xb, WkT, Kf, M, Dd, Dd);
  gemm_bf16_kernel<<<ggrid, 128, 0, stream>>>(Xb, WvT, Vf, M, Dd, Dd);

  rope_pack_kernel<<<dim3(Ss, Hh, Bb), 64, 0, stream>>>(Qf, Kf, Vf, Qb, Ktb, Vb);

  attn_kernel<<<dim3(Ss / 64, Hh, Bb), 128, 0, stream>>>(Qb, Ktb, Vb, Of);

  gemm_bf16_kernel<<<ggrid, 128, 0, stream>>>(Of, WoT, (float*)d_out, M, Dd, Dd);
}
